// PARC_reduced_14345190768961
// MI455X (gfx1250) — compile-verified, matrix operand order checked by reading, not run on hardware
//
#include <hip/hip_runtime.h>

// ---------------------------------------------------------------------------
// MeshGraphNet on MI455X (gfx1250), transposed-GEMM formulation:
//   H_out^T = W^T x H^T  via v_wmma_f32_16x16x32_f16
// Weights are pre-swizzled ONCE per launch into f16 A-fragment-layout blobs;
// per-block staging is a pure b128 copy into double-buffered LDS, overlapped
// with the current layer's WMMAs (one barrier per layer). Activations live
// entirely in registers across all 10 layers: the WMMA D layout maps
// lane-locally onto the next layer's B layout.
// ---------------------------------------------------------------------------

typedef __attribute__((ext_vector_type(16))) _Float16 v16h;
typedef __attribute__((ext_vector_type(8)))  float    v8f;

#define HID 64
#define NHIDL 9      // hidden layers after the first linear
#define NT 4
#define NF 2

__device__ __forceinline__ v8f wmma_f16(v16h A, v16h B, v8f C) {
    return __builtin_amdgcn_wmma_f32_16x16x32_f16(false, A, false, B,
                                                  (short)0, C, false, false);
}

// ---------------------------------------------------------------------------
// One-time weight swizzle: W (f32, [K][ldw]) -> f16 blob in A-fragment layout:
// blob[((kc*4+mt)*32 + l)*16 + h] = W[kc*32 + kmap(h, l>>4), mt*16 + (l&15)]
// gridDim.y = layer index (for stacked hidden-layer weights).
// ---------------------------------------------------------------------------
__global__ void swizzle_weights(const float* __restrict__ Ws, _Float16* __restrict__ blob,
                                int K, int kvalid, int ldw, int colw,
                                int srcStride, int dstStride)
{
    const int layer = blockIdx.y;
    const float* W = Ws + layer * srcStride;
    _Float16* dstb = blob + layer * dstStride;
    int i = blockIdx.x * blockDim.x + threadIdx.x;
    if (i < K * HID) {
        int h = i & 15;
        int l = (i >> 4) & 31;
        int fid = i >> 9;
        int kc = fid >> 2, mt = fid & 3;
        int k = kc * 32 + h + ((h & 8) ? 8 : 0) + ((l >> 4) * 8);
        int col = mt * 16 + (l & 15);
        float wv = (k < kvalid && col < colw) ? W[k * ldw + col] : 0.0f;
        dstb[i] = (_Float16)wv;
    }
}

// Vectorized copy of a pre-swizzled layer blob into LDS + bias stage.
template<int NKC>
__device__ __forceinline__ void stage_frag(
    _Float16* __restrict__ Wl, float* __restrict__ Bias,
    const _Float16* __restrict__ blob, const float* __restrict__ Bs,
    int colw, int tid)
{
    constexpr int N16B = NKC * 32 * HID / 8;   // 16-byte chunks
    const float4* s = (const float4*)blob;
    float4* d = (float4*)Wl;
#pragma unroll
    for (int j = 0; j < N16B / 128; ++j) d[j * 128 + tid] = s[j * 128 + tid];
    if (tid < HID) Bias[tid] = (tid < colw) ? Bs[tid] : 0.0f;
}

// One transposed layer: D[et][mt] = bias + sum_kc A(kc,mt) x B[et][kc]
// A fragments reused across both edge tiles (et).
template<int NKC>
__device__ __forceinline__ void layerT(
    const _Float16* __restrict__ Wl, const float* __restrict__ Bias,
    const v16h (&B)[2][NKC], v8f (&D)[2][4], int lane)
{
    const int laneHi8 = (lane >> 4) * 8;
#pragma unroll
    for (int mt = 0; mt < 4; ++mt) {
        v8f cb = *(const v8f*)&Bias[mt * 16 + laneHi8];
        v8f c0 = cb, c1 = cb;
#pragma unroll
        for (int kc = 0; kc < NKC; ++kc) {
            v16h A = *(const v16h*)&Wl[((kc * 4 + mt) * 32 + lane) * 16];
            c0 = wmma_f16(A, B[0][kc], c0);
            c1 = wmma_f16(A, B[1][kc], c1);
        }
        D[0][mt] = c0;
        D[1][mt] = c1;
    }
}

// Lane-local D -> next-layer B: B[kc][h] = cvt(act(D[2kc][h])), h<8
//                               B[kc][8+h] = cvt(act(D[2kc+1][h]))
__device__ __forceinline__ void d_to_b(const v8f (&D)[2][4], v16h (&B)[2][2], bool relu)
{
#pragma unroll
    for (int et = 0; et < 2; ++et)
#pragma unroll
        for (int kc = 0; kc < 2; ++kc) {
            v16h b;
#pragma unroll
            for (int j = 0; j < 8; ++j) {
                float v0 = D[et][2 * kc][j];
                float v1 = D[et][2 * kc + 1][j];
                if (relu) { v0 = fmaxf(v0, 0.0f); v1 = fmaxf(v1, 0.0f); }
                b[j]     = (_Float16)v0;
                b[8 + j] = (_Float16)v1;
            }
            B[et][kc] = b;
        }
}

// ---------------------------------------------------------------------------
// Edge MLP + scatter. Block = 128 threads = 4 waves; 32 edges/wave (2 B tiles)
// K0: padded first-layer input width (32 for mesh pass / xw=8, 128 for ds / xw=64)
// msg_in = [x[dst] | x[src]]; messages scattered by src.
// Weight staging double-buffered: stage layer l+1 while computing layer l.
// ---------------------------------------------------------------------------
template<int K0>
__global__ __launch_bounds__(128)
void edge_mlp_scatter(const float* __restrict__ x,
                      const int* __restrict__ srcIdx, const int* __restrict__ dstIdx,
                      const _Float16* __restrict__ wb0, const float* __restrict__ b0,
                      const _Float16* __restrict__ wbh, const float* __restrict__ bh,
                      float* __restrict__ node_acc, int E)
{
    constexpr int NKC0 = K0 / 32;
    constexpr int W0H = NKC0 * 32 * HID;                 // layer-0 blob halves
    constexpr int WBH = (W0H > 2 * 32 * HID) ? W0H : 2 * 32 * HID;
    __shared__ alignas(32) _Float16 Wbuf0[WBH];
    __shared__ alignas(32) _Float16 Wbuf1[2 * 32 * HID];
    __shared__ alignas(32) float BiasBuf[2][HID];

    const int tid = threadIdx.x;
    const int wave = tid >> 5;
    const int lane = tid & 31;
    const int laneLo = lane & 15;
    const int laneHi8 = (lane >> 4) * 8;
    const int eBase = blockIdx.x * 128 + wave * 32;

    // ---- Gather layer-0 B operand straight from global into registers ----
    v16h B0[2][NKC0];
#pragma unroll
    for (int et = 0; et < 2; ++et) {
        int e = eBase + et * 16 + laneLo;
        if constexpr (K0 == 128) {
            int nd = dstIdx[e], ns = srcIdx[e];
#pragma unroll
            for (int kc = 0; kc < 4; ++kc) {
                const float* xb = x + ((kc < 2) ? nd : ns) * 64 + (kc & 1) * 32 + laneHi8;
                v8f r0 = *(const v8f*)(xb);
                v8f r1 = *(const v8f*)(xb + 16);
                v16h b;
#pragma unroll
                for (int j = 0; j < 8; ++j) { b[j] = (_Float16)r0[j]; b[8 + j] = (_Float16)r1[j]; }
                B0[et][kc] = b;
            }
        } else {  // K0 == 32, mesh pass: xw = 8, features [dst(8) | src(8)], pad to 32
            int node = (laneHi8 != 0) ? srcIdx[e] : dstIdx[e];
            v8f r0 = *(const v8f*)(x + node * 8);
            v16h b;
#pragma unroll
            for (int j = 0; j < 8; ++j) { b[j] = (_Float16)r0[j]; b[8 + j] = (_Float16)0.0f; }
            B0[et][0] = b;
        }
    }

    // ---- Layer 0 (stage layer 1 into the other buffer while computing) ----
    stage_frag<NKC0>(Wbuf0, BiasBuf[0], wb0, b0, HID, tid);
    __syncthreads();
    stage_frag<2>(Wbuf1, BiasBuf[1], wbh, bh, HID, tid);
    __builtin_prefetch(wbh + HID * HID + tid * 32, 0, 3);
    v8f D[2][4];
    layerT<NKC0>(Wbuf0, BiasBuf[0], B0, D, lane);
    v16h Bh[2][2];
    d_to_b(D, Bh, /*relu*/true);
    __syncthreads();

    // ---- Hidden layers 1..9 (last without activation), double-buffered ----
    // hidden layer l resides in: l even -> Wbuf1 / BiasBuf[1], l odd -> Wbuf0 / BiasBuf[0]
#pragma unroll 1
    for (int l = 0; l < NHIDL; ++l) {
        const _Float16* Wc = (l & 1) ? Wbuf0 : Wbuf1;
        const float*    Bc = (l & 1) ? BiasBuf[0] : BiasBuf[1];
        _Float16*       Wn = (l & 1) ? Wbuf1 : Wbuf0;
        float*          Bn = (l & 1) ? BiasBuf[1] : BiasBuf[0];
        if (l + 1 < NHIDL) {
            stage_frag<2>(Wn, Bn, wbh + (l + 1) * HID * HID, bh + (l + 1) * HID, HID, tid);
            if (l + 2 < NHIDL)
                __builtin_prefetch(wbh + (l + 2) * HID * HID + tid * 32, 0, 3);
        }
        layerT<2>(Wc, Bc, Bh, D, lane);
        if (l < NHIDL - 1) d_to_b(D, Bh, /*relu*/true);
        __syncthreads();
    }

    // ---- Scatter-add messages from D registers, keyed by src ----
#pragma unroll
    for (int et = 0; et < 2; ++et) {
        int e = eBase + et * 16 + laneLo;
        float* nb = node_acc + srcIdx[e] * HID + laneHi8;
#pragma unroll
        for (int mt = 0; mt < 4; ++mt)
#pragma unroll
            for (int r = 0; r < 8; ++r)
                atomicAdd(nb + mt * 16 + r, D[et][mt][r]);
    }
}

// ---------------------------------------------------------------------------
// Node decoder: out[N,outw] = relu(acc @ w1 + b1) @ w2 + b2, transposed WMMA.
// Block = 128 threads, 32 nodes/wave, 128 nodes/block. Double-buffered weights.
// ---------------------------------------------------------------------------
__global__ __launch_bounds__(128)
void node_decoder(const float* __restrict__ acc,
                  const _Float16* __restrict__ wb1, const float* __restrict__ b1,
                  const _Float16* __restrict__ wb2, const float* __restrict__ b2,
                  int outw, float* __restrict__ out, int N)
{
    __shared__ alignas(32) _Float16 Wbuf0[HID * HID];
    __shared__ alignas(32) _Float16 Wbuf1[HID * HID];
    __shared__ alignas(32) float BiasBuf[2][HID];

    const int tid = threadIdx.x;
    const int wave = tid >> 5;
    const int lane = tid & 31;
    const int laneLo = lane & 15;
    const int laneHi8 = (lane >> 4) * 8;
    const int nBase = blockIdx.x * 128 + wave * 32;

    // Gather B from node accumulator
    v16h B[2][2];
#pragma unroll
    for (int et = 0; et < 2; ++et) {
        int n = nBase + et * 16 + laneLo;
        bool ok = (n < N);
#pragma unroll
        for (int kc = 0; kc < 2; ++kc) {
            v16h b;
            if (ok) {
                const float* ab = acc + n * HID + kc * 32 + laneHi8;
                v8f r0 = *(const v8f*)(ab);
                v8f r1 = *(const v8f*)(ab + 16);
#pragma unroll
                for (int j = 0; j < 8; ++j) { b[j] = (_Float16)r0[j]; b[8 + j] = (_Float16)r1[j]; }
            } else {
#pragma unroll
                for (int j = 0; j < 16; ++j) b[j] = (_Float16)0.0f;
            }
            B[et][kc] = b;
        }
    }

    // Layer 1: 64x64 + relu (stage layer 2 concurrently)
    stage_frag<2>(Wbuf0, BiasBuf[0], wb1, b1, HID, tid);
    __syncthreads();
    stage_frag<2>(Wbuf1, BiasBuf[1], wb2, b2, outw, tid);
    v8f D[2][4];
    layerT<2>(Wbuf0, BiasBuf[0], B, D, lane);
    d_to_b(D, B, /*relu*/true);
    __syncthreads();

    // Layer 2: 64 x outw (blob pre-padded with zero columns)
    layerT<2>(Wbuf1, BiasBuf[1], B, D, lane);

    // Store valid (node, feature) elements
#pragma unroll
    for (int et = 0; et < 2; ++et) {
        int n = nBase + et * 16 + laneLo;
        if (n < N) {
#pragma unroll
            for (int mt = 0; mt < 4; ++mt)
#pragma unroll
                for (int r = 0; r < 8; ++r) {
                    int f = mt * 16 + laneHi8 + r;
                    if (f < outw) out[n * outw + f] = D[et][mt][r];
                }
        }
    }
}

// ---------------------------------------------------------------------------
// Small scalar kernels
// ---------------------------------------------------------------------------
__global__ __launch_bounds__(64)
void node_encoder(const float* __restrict__ mesh_h, const float* __restrict__ F_prev,
                  const float* __restrict__ F_bnd, int t,
                  const float* __restrict__ w, const float* __restrict__ b,
                  float* __restrict__ h_out, int N)
{
    __shared__ float xin[35];
    const int n = blockIdx.x;
    const int o = threadIdx.x;
    if (o < 32)       xin[o] = mesh_h[n * 32 + o];
    else if (o < 34)  xin[o] = F_prev[n * NF + (o - 32)];
    else if (o == 34) xin[34] = F_bnd[n * NT + t];
    __syncthreads();
    float s = b[o];
#pragma unroll
    for (int k = 0; k < 35; ++k) s = fmaf(xin[k], w[k * HID + o], s);
    h_out[n * HID + o] = fmaxf(s, 0.0f);
}

__global__ void zero_kernel(float* __restrict__ p, int n) {
    int i = blockIdx.x * blockDim.x + threadIdx.x;
    if (i < n) p[i] = 0.0f;
}

__global__ void copy_kernel(float* __restrict__ dst, const float* __restrict__ src, int n) {
    int i = blockIdx.x * blockDim.x + threadIdx.x;
    if (i < n) dst[i] = src[i];
}

__global__ void update_kernel(const float* __restrict__ fdot, const float* __restrict__ dtp,
                              float* __restrict__ F_prev,
                              float* __restrict__ Fs_out, float* __restrict__ Fd_out,
                              int t, int N)
{
    int i = blockIdx.x * blockDim.x + threadIdx.x;
    if (i < N * NF) {
        int n = i >> 1, f = i & 1;
        float fd = fdot[i];
        float fn = F_prev[i] + dtp[0] * fd;
        Fs_out[n * ((NT - 1) * NF) + (t - 1) * NF + f] = fn;
        Fd_out[n * ((NT - 1) * NF) + (t - 1) * NF + f] = fd;
        F_prev[i] = fn;
    }
}

// ---------------------------------------------------------------------------
extern "C" void kernel_launch(void* const* d_in, const int* in_sizes, int n_in,
                              void* d_out, int out_size, void* d_ws, size_t ws_size,
                              hipStream_t stream)
{
    (void)in_sizes; (void)n_in; (void)out_size; (void)ws_size;
    const int N = 10000, E = 160000;

    const float* F_initial     = (const float*)d_in[0];
    const float* mesh_features = (const float*)d_in[1];
    const int*   edge_index    = (const int*)  d_in[2];
    const float* F_boundary    = (const float*)d_in[3];
    const float* dt            = (const float*)d_in[4];
    const float* md_edge_w0    = (const float*)d_in[5];
    const float* md_edge_b0    = (const float*)d_in[6];
    const float* md_edge_wh    = (const float*)d_in[7];
    const float* md_edge_bh    = (const float*)d_in[8];
    const float* md_dec_w1     = (const float*)d_in[9];
    const float* md_dec_b1     = (const float*)d_in[10];
    const float* md_dec_w2     = (const float*)d_in[11];
    const float* md_dec_b2     = (const float*)d_in[12];
    const float* ds_enc_w      = (const float*)d_in[13];
    const float* ds_enc_b      = (const float*)d_in[14];
    const float* ds_edge_w0    = (const float*)d_in[15];
    const float* ds_edge_b0    = (const float*)d_in[16];
    const float* ds_edge_wh    = (const float*)d_in[17];
    const float* ds_edge_bh    = (const float*)d_in[18];
    const float* ds_dec_w1     = (const float*)d_in[19];
    const float* ds_dec_b1     = (const float*)d_in[20];
    const float* ds_dec_w2     = (const float*)d_in[21];
    const float* ds_dec_b2     = (const float*)d_in[22];

    const int* src = edge_index;       // edge_index[0]
    const int* dst = edge_index + E;   // edge_index[1]

    float* ws       = (float*)d_ws;
    float* node_acc = ws;                   // N*64
    float* mesh_h   = node_acc + N * 64;    // N*32
    float* h_enc    = mesh_h + N * 32;      // N*64
    float* fdot     = h_enc + N * 64;       // N*2
    float* F_prev   = fdot + N * NF;        // N*2

    // f16 weight blobs (A-fragment layout), after float scratch
    _Float16* md0b = (_Float16*)(F_prev + N * NF);
    _Float16* mdhb = md0b + 32 * HID;        // 9 x 4096
    _Float16* ds0b = mdhb + 9 * HID * HID;   // 8192
    _Float16* dshb = ds0b + 128 * HID;       // 9 x 4096
    _Float16* m1b  = dshb + 9 * HID * HID;   // 4096
    _Float16* m2b  = m1b + HID * HID;
    _Float16* d1b  = m2b + HID * HID;
    _Float16* d2b  = d1b + HID * HID;

    float* Fs_out = (float*)d_out;               // [N, 3, 2]
    float* Fd_out = Fs_out + N * (NT - 1) * NF;  // [N, 3, 2]

    // ---- One-time weight swizzle into fragment-layout f16 blobs ----
    swizzle_weights<<<dim3(8, 1),  256, 0, stream>>>(md_edge_w0, md0b,  32,  16, HID, HID, 0, 0);
    swizzle_weights<<<dim3(16, 9), 256, 0, stream>>>(md_edge_wh, mdhb,  64,  64, HID, HID, HID * HID, HID * HID);
    swizzle_weights<<<dim3(32, 1), 256, 0, stream>>>(ds_edge_w0, ds0b, 128, 128, HID, HID, 0, 0);
    swizzle_weights<<<dim3(16, 9), 256, 0, stream>>>(ds_edge_wh, dshb,  64,  64, HID, HID, HID * HID, HID * HID);
    swizzle_weights<<<dim3(16, 1), 256, 0, stream>>>(md_dec_w1,  m1b,   64,  64, HID, HID, 0, 0);
    swizzle_weights<<<dim3(16, 1), 256, 0, stream>>>(md_dec_w2,  m2b,   64,  64, 32,  32,  0, 0);
    swizzle_weights<<<dim3(16, 1), 256, 0, stream>>>(ds_dec_w1,  d1b,   64,  64, HID, HID, 0, 0);
    swizzle_weights<<<dim3(16, 1), 256, 0, stream>>>(ds_dec_w2,  d2b,   64,  64, NF,  NF,  0, 0);

    // ---- Mesh descriptor pass ----
    zero_kernel<<<(N * 64 + 255) / 256, 256, 0, stream>>>(node_acc, N * 64);
    edge_mlp_scatter<32><<<E / 128, 128, 0, stream>>>(
        mesh_features, src, dst,
        md0b, md_edge_b0, mdhb, md_edge_bh, node_acc, E);
    node_decoder<<<(N + 127) / 128, 128, 0, stream>>>(
        node_acc, m1b, md_dec_b1, m2b, md_dec_b2, 32, mesh_h, N);

    copy_kernel<<<(N * NF + 255) / 256, 256, 0, stream>>>(F_prev, F_initial, N * NF);

    // ---- Timestep passes ----
    for (int t = 1; t < NT; ++t) {
        node_encoder<<<N, 64, 0, stream>>>(mesh_h, F_prev, F_boundary, t,
                                           ds_enc_w, ds_enc_b, h_enc, N);
        zero_kernel<<<(N * 64 + 255) / 256, 256, 0, stream>>>(node_acc, N * 64);
        edge_mlp_scatter<128><<<E / 128, 128, 0, stream>>>(
            h_enc, src, dst,
            ds0b, ds_edge_b0, dshb, ds_edge_bh, node_acc, E);
        node_decoder<<<(N + 127) / 128, 128, 0, stream>>>(
            node_acc, d1b, ds_dec_b1, d2b, ds_dec_b2, NF, fdot, N);
        update_kernel<<<(N * NF + 255) / 256, 256, 0, stream>>>(
            fdot, dt, F_prev, Fs_out, Fd_out, t, N);
    }
}
